// MeightWix_85014582657518
// MI455X (gfx1250) — compile-verified, matrix operand order checked by reading, not run on hardware
//
#include <hip/hip_runtime.h>

// ---------------------------------------------------------------------------
// Toeplitz GEMM for MI455X (gfx1250, wave32, WMMA + async-to-LDS).
//   out[m,n] = sum_k x[m,k] * v[(n+k) & 4095] + b[n]
//   M=16384, N=4096, K=4096, f32 in/out.  Memory-bound: ~512MB @ 23.3TB/s.
//
//  * bf16 hi/lo split, 3x V_WMMA_F32_16X16X32_BF16 per tile per K-chunk.
//  * v expanded once per block into LDS as dword-packed bf16 PAIRS so every
//    Toeplitz B-operand VGPR (K,K+1) is one aligned ds_load_b32.
//  * x tile (128 rows x 32 k, f32) staged through LDS with double-buffered
//    GLOBAL_LOAD_ASYNC_TO_LDS_B128 (ASYNCcnt) -- coalesced 128B segments,
//    fire-and-forget, overlapped with WMMA on the other buffer.
//  * 8 waves/block (4Mx2N), each wave owns 32x64 of C -> 24 WMMA / chunk.
// ---------------------------------------------------------------------------

typedef __attribute__((ext_vector_type(16))) __bf16 v16bf;
typedef __attribute__((ext_vector_type(8)))  float  v8f;

#define M_DIM 16384
#define N_DIM 4096
#define K_DIM 4096
#define KC    32
#define BM    128

union Frag {
    unsigned u[8];
    v16bf    v;
};

// LDS[lds_addr] = 16 bytes at gbase + voff  (async, tracked by ASYNCcnt)
static __device__ __forceinline__ void async_copy_b128(unsigned lds_addr,
                                                       const float* gbase,
                                                       unsigned voff_bytes) {
    asm volatile("global_load_async_to_lds_b128 %0, %1, %2 offset:0"
                 :: "v"(lds_addr), "v"(voff_bytes), "s"(gbase)
                 : "memory");
}
static __device__ __forceinline__ void wait_async_le4() {
    asm volatile("s_wait_asynccnt 0x4" ::: "memory");
}

__global__ __launch_bounds__(256)
void toeplitz_wmma_kernel(const float* __restrict__ X,
                          const float* __restrict__ V,
                          const float* __restrict__ Bias,
                          float* __restrict__ Out) {
    __shared__ unsigned pairs_hi[2 * K_DIM];      // 32 KB
    __shared__ unsigned pairs_lo[2 * K_DIM];      // 32 KB
    __shared__ float    Abuf[2][BM * KC];         // 2 x 16 KB double buffer

    const int tid = threadIdx.x;

    // ---- expand v into pair-packed bf16 hi/lo (once per block) ----
    for (int i = tid; i < 2 * K_DIM; i += 256) {
        float f0 = V[i & (K_DIM - 1)];
        float f1 = V[(i + 1) & (K_DIM - 1)];
        __bf16 h0 = (__bf16)f0, h1 = (__bf16)f1;
        __bf16 l0 = (__bf16)(f0 - (float)h0);
        __bf16 l1 = (__bf16)(f1 - (float)h1);
        pairs_hi[i] = (unsigned)__builtin_bit_cast(unsigned short, h0) |
                      ((unsigned)__builtin_bit_cast(unsigned short, h1) << 16);
        pairs_lo[i] = (unsigned)__builtin_bit_cast(unsigned short, l0) |
                      ((unsigned)__builtin_bit_cast(unsigned short, l1) << 16);
    }

    const int wid   = tid >> 5;
    const int lane  = tid & 31;
    const int half  = lane >> 4;
    const int l16   = lane & 15;
    const int waveM = wid & 3;                    // 4 waves along M
    const int waveN = wid >> 2;                   // 2 waves along N

    const int m0 = blockIdx.y * BM + waveM * 32;  // wave row base (32 rows)
    const int n0 = blockIdx.x * 128 + waveN * 64; // wave col base (64 cols)

    // Per-lane K-run start inside a 32-wide chunk (16-bit operand layout):
    // lane covers K in [kA,kA+7] and [16+kA,16+kA+7]; element e -> K=kA+e / 16+kA+(e-8).
    const int kA = 8 * half;

    // ---- async tile staging: thread copies 4 float4s of the 128x32 tile ----
    const float* gpanel = X + (size_t)blockIdx.y * BM * K_DIM;
    const unsigned lds_base = (unsigned)(size_t)(&Abuf[0][0]);
    auto issue_tile = [&](int kk, int nb) {
#pragma unroll
        for (int t = 0; t < 4; ++t) {
            const int f   = tid + t * 256;        // float4 index, 0..1023
            const int row = f >> 3;               // 8 float4 (=128B) per row
            const int c4  = f & 7;
            const unsigned voff  = (unsigned)((row * K_DIM + kk + c4 * 4) * 4);
            const unsigned laddr = lds_base + (unsigned)(nb * BM * KC * 4 + f * 16);
            async_copy_b128(laddr, gpanel, voff);
        }
    };

    issue_tile(0, 0);                             // prologue: chunk 0 -> buf 0
    __syncthreads();                              // pairs_* also ready

    v8f acc[2][4] = {};

    for (int c = 0; c < K_DIM / KC; ++c) {
        const int k0  = c * KC;
        const int buf = c & 1;

        // all waves finished reading buf^1 (chunk c-1) at previous barrier
        issue_tile((k0 + KC) & (K_DIM - 1), buf ^ 1);  // prefetch chunk c+1
        wait_async_le4();                              // my 4 copies of chunk c done
        __syncthreads();                               // everyone's copies done

        // ---- A fragments from staged LDS tile (ds_load_b128 x4 each) ----
        Frag ah[2], al[2];
#pragma unroll
        for (int mt = 0; mt < 2; ++mt) {
            const int rm = waveM * 32 + mt * 16 + l16;
            const float4* p = (const float4*)(&Abuf[buf][rm * KC]);
            float4 r0 = p[kA / 4 + 0];
            float4 r1 = p[kA / 4 + 1];
            float4 r2 = p[kA / 4 + 4];
            float4 r3 = p[kA / 4 + 5];
            float fa[16] = { r0.x, r0.y, r0.z, r0.w,  r1.x, r1.y, r1.z, r1.w,
                             r2.x, r2.y, r2.z, r2.w,  r3.x, r3.y, r3.z, r3.w };
#pragma unroll
            for (int e = 0; e < 16; ++e) {
                __bf16 h = (__bf16)fa[e];              // v_cvt_pk_bf16_f32
                ah[mt].v[e] = h;
                al[mt].v[e] = (__bf16)(fa[e] - (float)h);
            }
        }

        // ---- B fragments: Toeplitz column n = n0 + nt*16 + l16 ----
        Frag bh[4], bl[4];
#pragma unroll
        for (int nt = 0; nt < 4; ++nt) {
            const int base = k0 + n0 + nt * 16 + l16;  // fau index at K=0
#pragma unroll
            for (int q = 0; q < 8; ++q) {
                const int kk = (q < 4) ? (kA + 2 * q) : (16 + kA + 2 * (q - 4));
                bh[nt].u[q] = pairs_hi[base + kk];
                bl[nt].u[q] = pairs_lo[base + kk];
            }
        }

        // ---- 24x WMMA: hi*hi + hi*lo + lo*hi ----
#pragma unroll
        for (int mt = 0; mt < 2; ++mt) {
#pragma unroll
            for (int nt = 0; nt < 4; ++nt) {
                acc[mt][nt] = __builtin_amdgcn_wmma_f32_16x16x32_bf16(
                    false, ah[mt].v, false, bh[nt].v, (short)0, acc[mt][nt], false, false);
                acc[mt][nt] = __builtin_amdgcn_wmma_f32_16x16x32_bf16(
                    false, ah[mt].v, false, bl[nt].v, (short)0, acc[mt][nt], false, false);
                acc[mt][nt] = __builtin_amdgcn_wmma_f32_16x16x32_bf16(
                    false, al[mt].v, false, bh[nt].v, (short)0, acc[mt][nt], false, false);
            }
        }

        __syncthreads();   // done reading buf before it is refilled next iter
    }

    // ---- Epilogue: bias + store (C/D layout: VGPR r -> row r+8*half, col l16)
#pragma unroll
    for (int nt = 0; nt < 4; ++nt) {
        const float bias = Bias[n0 + nt * 16 + l16];
#pragma unroll
        for (int mt = 0; mt < 2; ++mt) {
            float* orow = Out + (size_t)(m0 + mt * 16 + 8 * half) * N_DIM
                              + n0 + nt * 16 + l16;
#pragma unroll
            for (int r = 0; r < 8; ++r) {
                orow[(size_t)r * N_DIM] = acc[mt][nt][r] + bias;
            }
        }
    }
}

extern "C" void kernel_launch(void* const* d_in, const int* in_sizes, int n_in,
                              void* d_out, int out_size, void* d_ws, size_t ws_size,
                              hipStream_t stream) {
    const float* x = (const float*)d_in[0];   // (16384, 4096)
    const float* v = (const float*)d_in[1];   // (4096,)
    const float* b = (const float*)d_in[2];   // (4096,)
    float* out = (float*)d_out;               // (16384, 4096)

    dim3 grid(N_DIM / 128, M_DIM / 128);      // (32, 128)
    toeplitz_wmma_kernel<<<grid, 256, 0, stream>>>(x, v, b, out);
}
